// Simple_6270652252303
// MI455X (gfx1250) — matrix operand order repaired and checked by reading, not compiled, not run on hardware
//
#include <hip/hip_runtime.h>
#include <hip/hip_bf16.h>

// ---------------- WMMA types ----------------
typedef __attribute__((ext_vector_type(16))) _Float16 v16h;
typedef __attribute__((ext_vector_type(8)))  _Float16 v8h;
typedef __attribute__((ext_vector_type(8)))  float    v8f;

#define TOK   4032      // B*L = 64*63
#define BB    64
#define LL    63
#define DD    100
#define NN    300
#define KPAD  128       // all GEMM K dims (100, 128) padded to 128
#define KCH   (KPAD/32) // 4 K-chunks
#define SCAN_S 8        // n-dimension split in scan block (800 = 8*100 threads)

// =====================================================================
// Pack activations: X (M x K f32, row-major) -> Xp (M x KPAD f16), zero pad.
// =====================================================================
__global__ __launch_bounds__(256) void pack_x_f16(
    const float* __restrict__ X, _Float16* __restrict__ Xp, int M, int K)
{
    const int i = blockIdx.x * blockDim.x + threadIdx.x;
    if (i >= M * KPAD) return;
    const int m = i / KPAD, k = i - m * KPAD;
    Xp[i] = (k < K) ? (_Float16)X[(size_t)m * K + k] : (_Float16)0.f;
}

// =====================================================================
// Pack weights W (K x N f32) into WMMA B-fragment order (zero padded):
//   Wp[((nTile*KCH + kk)*32 + lane)*16 + e]
//   lane: g = lane>>4, col n = nTile*16 + (lane&15); element e -> k = kk*32 + g*8 + (e&7) + (e>>3)*16
// =====================================================================
__global__ __launch_bounds__(256) void pack_w_frag(
    const float* __restrict__ W, _Float16* __restrict__ Wp,
    int K, int N, int total)
{
    const int idx = blockIdx.x * blockDim.x + threadIdx.x;
    if (idx >= total) return;
    const int e    = idx & 15;
    const int lane = (idx >> 4) & 31;
    const int frag = idx >> 9;            // nTile*KCH + kk
    const int nTile = frag / KCH;
    const int kk    = frag - nTile * KCH;
    const int g  = lane >> 4;
    const int n  = nTile * 16 + (lane & 15);
    const int k  = (kk << 5) + g * 8 + (e & 7) + ((e >> 3) << 4);
    Wp[idx] = (k < K && n < N) ? (_Float16)W[(size_t)k * N + n] : (_Float16)0.f;
}

// =====================================================================
// Wave-level WMMA GEMM on packed f16 operands: Y = X @ W + bias (f32 out).
// One wave per 16x16 tile; 4x(2 b128 A loads + 1 v16h B load + v_wmma).
// =====================================================================
__global__ __launch_bounds__(32) void gemm_wmma_packed(
    const _Float16* __restrict__ Xp,   // M x KPAD row-major f16
    const _Float16* __restrict__ Wp,   // fragment-packed f16
    const float* __restrict__ bias,    // N
    float* __restrict__ Y,             // M x N f32
    int M, int N)
{
    const int lane  = threadIdx.x & 31;
    const int g     = lane >> 4;
    const int lr    = lane & 15;
    const int mTile = blockIdx.x;
    const int nTile = blockIdx.y;

    const _Float16* xrow = Xp + (size_t)(mTile * 16 + lr) * KPAD;
    const v16h*     Wpv  = (const v16h*)Wp;

    v8f acc = {};
    #pragma unroll
    for (int kk = 0; kk < KCH; ++kk) {
        const v8h a0 = *(const v8h*)(xrow + kk * 32 + g * 8);
        const v8h a1 = *(const v8h*)(xrow + kk * 32 + 16 + g * 8);
        v16h a;
        #pragma unroll
        for (int e = 0; e < 8; ++e) { a[e] = a0[e]; a[8 + e] = a1[e]; }
        const v16h b = Wpv[(size_t)(nTile * KCH + kk) * 32 + lane];
        acc = __builtin_amdgcn_wmma_f32_16x16x32_f16(
                  false, a, false, b, (short)0, acc, false, false);
    }

    const int n = nTile * 16 + lr;
    if (n < N) {
        const float bv = bias[n];
        #pragma unroll
        for (int r = 0; r < 8; ++r) {
            const int mm = mTile * 16 + r + g * 8;   // D: VGPR r -> row r + g*8
            Y[(size_t)mm * N + n] = acc[r] + bv;
        }
    }
}

// =====================================================================
// In-place LayerNorm(last dim) + LeakyReLU(0.01). One wave per row.
// =====================================================================
__global__ __launch_bounds__(256) void ln_lrelu_kernel(
    float* __restrict__ Y, const float* __restrict__ gamma,
    const float* __restrict__ beta, int N, int rows)
{
    const int row = blockIdx.x * (blockDim.x >> 5) + (threadIdx.x >> 5);
    if (row >= rows) return;
    const int lane = threadIdx.x & 31;
    float* y = Y + (size_t)row * N;

    float s = 0.f, sq = 0.f;
    for (int i = lane; i < N; i += 32) { float v = y[i]; s += v; sq += v * v; }
    #pragma unroll
    for (int off = 16; off > 0; off >>= 1) {
        s  += __shfl_xor(s,  off, 32);
        sq += __shfl_xor(sq, off, 32);
    }
    const float mean = s / (float)N;
    const float var  = sq / (float)N - mean * mean;
    const float inv  = rsqrtf(var + 1e-5f);
    for (int i = lane; i < N; i += 32) {
        float v = (y[i] - mean) * inv * gamma[i] + beta[i];
        y[i] = (v >= 0.f) ? v : 0.01f * v;
    }
}

// =====================================================================
// Fused S6 selective scan, double-buffered staging. One block per batch b.
// Thread tid = s*100 + d owns h[d, n] (n = s + 8j, j=0..37) in REGISTERS
// across all 63 steps; A row register-resident. Step t+1's vectors are
// global-loaded into registers BEFORE step t's compute, then committed to
// the ping-pong LDS buffer, hiding L2 latency behind the exp-heavy update.
// =====================================================================
__global__ __launch_bounds__(SCAN_S * DD) void s6_scan_kernel(
    const float* __restrict__ Xin,      // (TOK, 100)
    const float* __restrict__ DeltaRaw, // (TOK, 100)  pre-softplus
    const float* __restrict__ Bm,       // (TOK, 300)
    const float* __restrict__ Cm,       // (TOK, 300)
    const float* __restrict__ A,        // (100, 300)
    float* __restrict__ Yout)           // (TOK, 100)
{
    __shared__ float sx[2][DD], sdl[2][DD], sB[2][NN], sC[2][NN];
    __shared__ float ypart[SCAN_S][DD];

    const int tid = threadIdx.x;
    const int d   = tid % DD;
    const int s   = tid / DD;          // 0..7
    const int b   = blockIdx.x;
    const size_t base = (size_t)b * LL;

    const int JMAX = (NN + SCAN_S - 1) / SCAN_S;   // 38
    float h[JMAX], av[JMAX];
    #pragma unroll
    for (int j = 0; j < JMAX; ++j) {
        const int n = s + SCAN_S * j;
        h[j]  = 0.f;
        av[j] = (n < NN) ? A[d * NN + n] : 0.f;
    }

    // prime buffer 0 with step 0
    if (tid < DD) {
        sx[0][tid] = Xin[base * DD + tid];
        float dr = DeltaRaw[base * DD + tid];
        sdl[0][tid] = (dr > 20.f) ? dr : log1pf(__expf(dr));   // softplus
    }
    if (tid < NN) {
        sB[0][tid] = Bm[base * NN + tid];
        sC[0][tid] = Cm[base * NN + tid];
    }
    __syncthreads();

    for (int t = 0; t < LL; ++t) {
        const int cur = t & 1, nxt = cur ^ 1;

        // ---- issue step t+1 loads (latency overlapped with compute) ----
        float nx = 0.f, ndr = 0.f, nB = 0.f, nC = 0.f;
        if (t + 1 < LL) {
            const size_t tok1 = base + t + 1;
            if (tid < DD) {
                nx  = Xin[tok1 * DD + tid];
                ndr = DeltaRaw[tok1 * DD + tid];
            }
            if (tid < NN) {
                nB = Bm[tok1 * NN + tid];
                nC = Cm[tok1 * NN + tid];
            }
        }

        // ---- compute step t from current buffer ----
        const float dl  = sdl[cur][d];
        const float dlx = dl * sx[cur][d];
        float acc = 0.f;
        #pragma unroll
        for (int j = 0; j < JMAX; ++j) {
            const int n = s + SCAN_S * j;
            if (n < NN) {
                h[j] = __expf(dl * av[j]) * h[j] + dlx * sB[cur][n];
                acc += sC[cur][n] * h[j];
            }
        }
        ypart[s][d] = acc;

        // ---- commit prefetched step t+1 into the other buffer ----
        if (t + 1 < LL) {
            if (tid < DD) {
                sx[nxt][tid]  = nx;
                sdl[nxt][tid] = (ndr > 20.f) ? ndr : log1pf(__expf(ndr));
            }
            if (tid < NN) {
                sB[nxt][tid] = nB;
                sC[nxt][tid] = nC;
            }
        }
        __syncthreads();

        if (tid < DD) {
            float y = 0.f;
            #pragma unroll
            for (int ss = 0; ss < SCAN_S; ++ss) y += ypart[ss][tid];
            Yout[(base + t) * DD + tid] = y;
        }
        __syncthreads();   // ypart reused & nxt buffer read next iteration
    }
}

// =====================================================================
// Final tiny layer: out(tok,c) = cb2[c] + sum_{j<50} O1[tok,j]*cW2[j,c]
// =====================================================================
__global__ __launch_bounds__(256) void head2_kernel(
    const float* __restrict__ O1, const float* __restrict__ W2,
    const float* __restrict__ b2, float* __restrict__ Out, int total)
{
    const int i = blockIdx.x * blockDim.x + threadIdx.x;
    if (i >= total) return;
    const int tok = i >> 1, c = i & 1;
    const float* o = O1 + (size_t)tok * 50;
    float s = b2[c];
    #pragma unroll
    for (int j = 0; j < 50; ++j) s += o[j] * W2[j * 2 + c];
    Out[i] = s;
}

// =====================================================================
extern "C" void kernel_launch(void* const* d_in, const int* in_sizes, int n_in,
                              void* d_out, int out_size, void* d_ws, size_t ws_size,
                              hipStream_t stream)
{
    const float* text = (const float*)d_in[0];
    const float* tW1  = (const float*)d_in[3];
    const float* tb1  = (const float*)d_in[4];
    const float* tg1  = (const float*)d_in[5];
    const float* tbe1 = (const float*)d_in[6];
    const float* tW2  = (const float*)d_in[7];
    const float* tb2  = (const float*)d_in[8];
    const float* tg2  = (const float*)d_in[9];
    const float* tbe2 = (const float*)d_in[10];
    const float* sW1  = (const float*)d_in[11];
    const float* sb1  = (const float*)d_in[12];
    const float* sW2  = (const float*)d_in[13];
    const float* sb2  = (const float*)d_in[14];
    const float* sW3  = (const float*)d_in[15];
    const float* sb3  = (const float*)d_in[16];
    const float* sA   = (const float*)d_in[17];
    // d_in[18..33]: audio/video branch params -- dead code (never reach output)
    const float* cW1  = (const float*)d_in[34];
    const float* cb1  = (const float*)d_in[35];
    const float* cW2  = (const float*)d_in[36];
    const float* cb2  = (const float*)d_in[37];
    float* out = (float*)d_out;

    // ---- workspace layout: f16 packed region first (32B aligned), then f32 ----
    _Float16* Xp = (_Float16*)d_ws;                 // TOK*KPAD = 516096 f16
    _Float16* Wp = Xp + (size_t)TOK * KPAD;         // up to 38912 f16 (pad to 40960)
    float* fb = (float*)(Wp + 40960);
    float* t1 = fb;                    // (TOK,128)
    float* t2 = t1 + (size_t)TOK*128;  // (TOK,100)
    float* P1 = t2 + (size_t)TOK*100;  // delta raw (TOK,100)
    float* P2 = P1 + (size_t)TOK*100;  // Bm (TOK,300)
    float* P3 = P2 + (size_t)TOK*300;  // Cm (TOK,300)
    float* Ys = P3 + (size_t)TOK*300;  // scan out (TOK,100)
    float* O1 = Ys + (size_t)TOK*100;  // head o1 (TOK,50)

    const dim3 blk32(32), blk256(256);
    const int MT = TOK / 16;                        // 252 M-tiles
    const int gx = (TOK * KPAD + 255) / 256;        // pack_x grid

    auto wtotal = [](int N) { return ((N + 15) / 16) * KCH * 512; };

    // ---- text layer 1: (TOKx100)@(100x128)+b -> LN -> LReLU ----
    pack_x_f16<<<dim3(gx), blk256, 0, stream>>>(text, Xp, TOK, 100);
    { int tw = wtotal(128);
      pack_w_frag<<<dim3((tw+255)/256), blk256, 0, stream>>>(tW1, Wp, 100, 128, tw); }
    gemm_wmma_packed<<<dim3(MT, 8), blk32, 0, stream>>>(Xp, Wp, tb1, t1, TOK, 128);
    ln_lrelu_kernel<<<dim3(TOK/8), blk256, 0, stream>>>(t1, tg1, tbe1, 128, TOK);

    // ---- text layer 2: (TOKx128)@(128x100)+b -> LN -> LReLU ----
    pack_x_f16<<<dim3(gx), blk256, 0, stream>>>(t1, Xp, TOK, 128);
    { int tw = wtotal(100);
      pack_w_frag<<<dim3((tw+255)/256), blk256, 0, stream>>>(tW2, Wp, 128, 100, tw); }
    gemm_wmma_packed<<<dim3(MT, 7), blk32, 0, stream>>>(Xp, Wp, tb2, t2, TOK, 100);
    ln_lrelu_kernel<<<dim3(TOK/8), blk256, 0, stream>>>(t2, tg2, tbe2, 100, TOK);

    // ---- S6 projections: pack t2 once, three weight packs + GEMMs ----
    pack_x_f16<<<dim3(gx), blk256, 0, stream>>>(t2, Xp, TOK, 100);
    { int tw = wtotal(100);
      pack_w_frag<<<dim3((tw+255)/256), blk256, 0, stream>>>(sW1, Wp, 100, 100, tw); }
    gemm_wmma_packed<<<dim3(MT, 7), blk32, 0, stream>>>(Xp, Wp, sb1, P1, TOK, 100);
    { int tw = wtotal(300);
      pack_w_frag<<<dim3((tw+255)/256), blk256, 0, stream>>>(sW2, Wp, 100, 300, tw); }
    gemm_wmma_packed<<<dim3(MT, 19), blk32, 0, stream>>>(Xp, Wp, sb2, P2, TOK, 300);
    { int tw = wtotal(300);
      pack_w_frag<<<dim3((tw+255)/256), blk256, 0, stream>>>(sW3, Wp, 100, 300, tw); }
    gemm_wmma_packed<<<dim3(MT, 19), blk32, 0, stream>>>(Xp, Wp, sb3, P3, TOK, 300);

    // ---- fused selective scan (register-resident state, double-buffered) ----
    s6_scan_kernel<<<dim3(BB), dim3(SCAN_S * DD), 0, stream>>>(t2, P1, P2, P3, sA, Ys);

    // ---- head: (TOKx100)@(100x50)+b via WMMA, then 50->2 ----
    pack_x_f16<<<dim3(gx), blk256, 0, stream>>>(Ys, Xp, TOK, 100);
    { int tw = wtotal(50);
      pack_w_frag<<<dim3((tw+255)/256), blk256, 0, stream>>>(cW1, Wp, 100, 50, tw); }
    gemm_wmma_packed<<<dim3(MT, 4), blk32, 0, stream>>>(Xp, Wp, cb1, O1, TOK, 50);
    head2_kernel<<<dim3((TOK*2 + 255)/256), blk256, 0, stream>>>(O1, cW2, cb2, out, TOK*2);
}
